// EncoderMultiHeadAttention_7541962572325
// MI455X (gfx1250) — compile-verified, matrix operand order checked by reading, not run on hardware
//
#include <hip/hip_runtime.h>
#include <hip/hip_bf16.h>

typedef __attribute__((ext_vector_type(16))) _Float16 v16h;
typedef __attribute__((ext_vector_type(8)))  _Float16 v8h;
typedef __attribute__((ext_vector_type(4)))  _Float16 v4h;
typedef __attribute__((ext_vector_type(8)))  float    v8f;

#define DEV __device__ __forceinline__

static constexpr int Bn  = 4;
static constexpr int Sn  = 2048;
static constexpr int Dn  = 1024;
static constexpr int Hn  = 16;
static constexpr int DKn = 64;
static constexpr int Mn  = Bn * Sn;          // 8192 rows total

#if defined(__has_builtin)
#  if __has_builtin(__builtin_amdgcn_tensor_load_to_lds)
#    define HAVE_TDM 1
#  endif
#endif
#ifndef HAVE_TDM
#  define HAVE_TDM 0
#endif

// Load a 16-half WMMA fragment (A-layout or B-layout, same per-lane pattern):
// lanes 0-15 take K {0..7} and {16..23}, lanes 16-31 take K {8..15},{24..31}.
// Caller bakes the +8 lane offset into `p`.
DEV v16h ldfrag(const _Float16* p) {
    union { v16h v; v8h h[2]; } u;
    u.h[0] = *(const v8h*)(p);
    u.h[1] = *(const v8h*)(p + 16);
    return u.v;
}

#if HAVE_TDM
typedef __attribute__((ext_vector_type(4))) unsigned int v4u;
typedef __attribute__((ext_vector_type(8))) int          v8i_;
typedef __attribute__((ext_vector_type(4))) int          v4i_;

// 2D f16 tile load, Global -> LDS via the Tensor Data Mover.
// D# group0: count=1, lds_addr, 57-bit global addr, type=2 ("image").
// D# group1: data_size=2B, optional LDS row padding (pad_code: rows of
//            2^(pad_code+1) dwords get +4 dwords of padding; 0 = no padding),
//            tensor/tile dims and dim0 stride (elements).
DEV void tdm_load_2d_f16(unsigned lds_off, const _Float16* gptr,
                         unsigned td0, unsigned td1,
                         unsigned tile0, unsigned tile1,
                         unsigned stride0, unsigned pad_code) {
    unsigned long long ga = (unsigned long long)(const void*)gptr;
    v4u g0;
    g0[0] = 1u;                                   // count = 1 valid descriptor
    g0[1] = lds_off;                              // LDS byte address
    g0[2] = (unsigned)(ga & 0xffffffffu);
    g0[3] = (unsigned)((ga >> 32) & 0x01ffffffu) | (2u << 30);  // type=2
    unsigned w0 = (1u << 16);                     // data_size = 2 bytes
    if (pad_code)                                 // +4 dwords per 2^(pad_code+1) dwords
        w0 |= (1u << 20) | (pad_code << 22) | (3u << 25);
    v8i_ g1;
    g1[0] = (int)w0;
    g1[1] = (int)((td0 & 0xffffu) << 16);                         // dim0 lo
    g1[2] = (int)(((td0 >> 16) & 0xffffu) | ((td1 & 0xffffu) << 16));
    g1[3] = (int)(((td1 >> 16) & 0xffffu) | ((tile0 & 0xffffu) << 16));
    g1[4] = (int)(tile1 & 0xffffu);               // tile_dim1 (tile_dim2 = 0)
    g1[5] = (int)stride0;                         // dim0 stride lo32
    g1[6] = 0;
    g1[7] = 0;
    v4i_ z4 = {0, 0, 0, 0};
#if __clang_major__ >= 23
    v8i_ z8 = {0, 0, 0, 0, 0, 0, 0, 0};
    __builtin_amdgcn_tensor_load_to_lds(g0, g1, z4, z4, z8, 0);
#else
    __builtin_amdgcn_tensor_load_to_lds(g0, g1, z4, z4, 0);
#endif
}
#endif

// ---------------------------------------------------------------------------
// 1) Row LayerNorm fp32 -> f16.  One block (256 thr) per row of D=1024.
// ---------------------------------------------------------------------------
__global__ __launch_bounds__(256) void ln_rows_f16(
        const float* __restrict__ x, const float* __restrict__ g,
        const float* __restrict__ bb, _Float16* __restrict__ y) {
    __shared__ float red[8];
    const int row  = blockIdx.x;
    const int lane = threadIdx.x & 31;
    const int wave = threadIdx.x >> 5;

    const float4* xr = (const float4*)(x + (size_t)row * Dn);
    float4 vv = xr[threadIdx.x];

    float s = vv.x + vv.y + vv.z + vv.w;
    #pragma unroll
    for (int m = 16; m; m >>= 1) s += __shfl_xor(s, m, 32);
    if (!lane) red[wave] = s;
    __syncthreads();
    float tot = 0.f;
    #pragma unroll
    for (int i = 0; i < 8; ++i) tot += red[i];
    const float mu = tot * (1.0f / Dn);

    const float d0 = vv.x - mu, d1 = vv.y - mu, d2 = vv.z - mu, d3 = vv.w - mu;
    float s2 = d0*d0 + d1*d1 + d2*d2 + d3*d3;
    #pragma unroll
    for (int m = 16; m; m >>= 1) s2 += __shfl_xor(s2, m, 32);
    __syncthreads();
    if (!lane) red[wave] = s2;
    __syncthreads();
    float tv = 0.f;
    #pragma unroll
    for (int i = 0; i < 8; ++i) tv += red[i];
    const float inv = rsqrtf(tv * (1.0f / Dn) + 1e-5f);

    const float4 gg = ((const float4*)g)[threadIdx.x];
    const float4 bv = ((const float4*)bb)[threadIdx.x];
    v4h o;
    o[0] = (_Float16)(d0 * inv * gg.x + bv.x);
    o[1] = (_Float16)(d1 * inv * gg.y + bv.y);
    o[2] = (_Float16)(d2 * inv * gg.z + bv.z);
    o[3] = (_Float16)(d3 * inv * gg.w + bv.w);
    ((v4h*)(y + (size_t)row * Dn))[threadIdx.x] = o;
}

// ---------------------------------------------------------------------------
// 2) Weight prep: fp32 W[K][N] -> f16 Wt[N][K].
// ---------------------------------------------------------------------------
__global__ __launch_bounds__(256) void w_to_f16_t(
        const float* __restrict__ W, _Float16* __restrict__ Wt) {
    const int idx = blockIdx.x * 256 + threadIdx.x;   // over Dn*Dn
    const int n = idx >> 10;          // / Dn
    const int k = idx & (Dn - 1);
    Wt[idx] = (_Float16)W[(size_t)k * Dn + n];
}

// ---------------------------------------------------------------------------
// 3) WMMA GEMM: C[M,N] = A[M,K](f16) * Bt[N,K](f16).
//    B panel (64 x 128 per step) staged in LDS by the TDM (padded rows ->
//    conflict-free ds_load_b128 for all 8 waves).  A streamed from global.
//    MODE 0: f16 out scattered to [B,H,S,dk]
//    MODE 1: f16 out scattered to [B,H,dk,S]   (V transposed for PV stage)
//    MODE 2: f32 out [M,N] + residual (final Wo projection)
// ---------------------------------------------------------------------------
static constexpr int RS = 136;   // padded LDS row stride in halves (272 B)

template<int MODE>
__global__ __launch_bounds__(256) void wmma_gemm(
        const _Float16* __restrict__ A, const _Float16* __restrict__ Bt,
        void* __restrict__ Y, const float* __restrict__ resid) {
    constexpr int Kd = Dn;
    __shared__ _Float16 tileB[64 * RS];          // ~17 KB
    const int lane = threadIdx.x & 31;
    const int wave = threadIdx.x >> 5;
    const int m0 = blockIdx.y * 128 + wave * 16;
    const int n0 = blockIdx.x * 64;
    const int off = (lane >= 16) ? 8 : 0;
    const int r16 = lane & 15;

    v8f acc0 = {}, acc1 = {}, acc2 = {}, acc3 = {};
    const _Float16* ap = A + (size_t)(m0 + r16) * Kd + off;

    for (int kO = 0; kO < Kd; kO += 128) {
#if HAVE_TDM
        if (wave == 0) {
            tdm_load_2d_f16((unsigned)(size_t)(void*)&tileB[0],
                            Bt + (size_t)n0 * Kd + kO,
                            /*td0=*/128, /*td1=*/64,
                            /*tile0=*/128, /*tile1=*/64,
                            /*stride0=*/Kd, /*pad_code=*/5);   // 64-dw rows
            __builtin_amdgcn_s_wait_tensorcnt(0);
        }
#else
        for (int t = threadIdx.x; t < 64 * 16; t += 256) {
            const int rrow = t >> 4;
            const int cc = (t & 15) * 8;
            *(v8h*)&tileB[rrow * RS + cc] =
                *(const v8h*)&Bt[(size_t)(n0 + rrow) * Kd + kO + cc];
        }
#endif
        __syncthreads();
        if (kO + 128 < Kd) __builtin_prefetch(ap + kO + 128, 0, 1);
        #pragma unroll
        for (int kk = 0; kk < 128; kk += 32) {
            v16h a  = ldfrag(ap + kO + kk);
            v16h b0 = ldfrag(&tileB[( 0 + r16) * RS + kk + off]);
            v16h b1 = ldfrag(&tileB[(16 + r16) * RS + kk + off]);
            v16h b2 = ldfrag(&tileB[(32 + r16) * RS + kk + off]);
            v16h b3 = ldfrag(&tileB[(48 + r16) * RS + kk + off]);
            acc0 = __builtin_amdgcn_wmma_f32_16x16x32_f16(false, a, false, b0, (short)0, acc0, false, false);
            acc1 = __builtin_amdgcn_wmma_f32_16x16x32_f16(false, a, false, b1, (short)0, acc1, false, false);
            acc2 = __builtin_amdgcn_wmma_f32_16x16x32_f16(false, a, false, b2, (short)0, acc2, false, false);
            acc3 = __builtin_amdgcn_wmma_f32_16x16x32_f16(false, a, false, b3, (short)0, acc3, false, false);
        }
        __syncthreads();   // protect tileB against next chunk's overwrite
    }

    const int rowb = (lane >= 16) ? 8 : 0;
    v8f accs[4] = {acc0, acc1, acc2, acc3};
    #pragma unroll
    for (int t = 0; t < 4; ++t) {
        #pragma unroll
        for (int rr = 0; rr < 8; ++rr) {
            const int mm = m0 + rowb + rr;
            const int nn = n0 + t * 16 + r16;
            const float val = accs[t][rr];
            if constexpr (MODE == 2) {
                ((float*)Y)[(size_t)mm * Dn + nn] = val + resid[(size_t)mm * Dn + nn];
            } else {
                const int bidx = mm >> 11;           // / Sn
                const int ss   = mm & (Sn - 1);
                const int hh   = nn >> 6;            // / DKn
                const int dd   = nn & (DKn - 1);
                size_t idx;
                if constexpr (MODE == 0)
                    idx = (((size_t)bidx * Hn + hh) * Sn + ss) * DKn + dd;
                else
                    idx = (((size_t)bidx * Hn + hh) * DKn + dd) * Sn + ss;
                ((_Float16*)Y)[idx] = (_Float16)val;
            }
        }
    }
}

// ---------------------------------------------------------------------------
// 4) Flash attention.  All 8 waves of a block share one (b,h) and 8
//    consecutive 16-row q-tiles, so K/V panels (128 kv columns) are staged
//    once per block in LDS by the TDM and consumed by every wave (8x less
//    global traffic).  Row padding keeps fragment ds reads conflict-free.
//    Online softmax; P converted C-layout -> A-fragment layout via per-wave
//    LDS; row sums come from a P x ones WMMA.
// ---------------------------------------------------------------------------
static constexpr int KRS = 72;    // K tile row stride (halves): 36-dw rows
static constexpr int VRS = 136;   // V tile row stride (halves): 68-dw rows

__global__ __launch_bounds__(256) void flash_attn_kernel(
        const _Float16* __restrict__ Qh,   // [B,H,S,dk]
        const _Float16* __restrict__ Kh,   // [B,H,S,dk]
        const _Float16* __restrict__ VT,   // [B,H,dk,S]
        const int*      __restrict__ mask, // [B,S]
        _Float16*       __restrict__ Oa) { // [B*S, D]
    __shared__ _Float16 Kt[128 * KRS];     // 18 KB: 128 kv rows x 64 dk
    __shared__ _Float16 Vt[64 * VRS];      // 17 KB: 64 dk rows x 128 kv
    __shared__ _Float16 Pl[8][16 * 32];    // 8 KB per-wave P staging
    const int lane = threadIdx.x & 31;
    const int wave = threadIdx.x >> 5;
    const int gw = blockIdx.x * 8 + wave;
    const int qt = gw & (Sn / 16 - 1);
    const int h  = (gw >> 7) & (Hn - 1);
    const int b  = gw >> 11;
    const int q0 = qt * 16;
    const int off  = (lane >= 16) ? 8 : 0;
    const int r16  = lane & 15;
    const int rowb = (lane >= 16) ? 8 : 0;

    const size_t bh = (size_t)b * Hn + h;
    const _Float16* Qbase = Qh + (bh * Sn + q0 + r16) * DKn + off;
    const v16h qf0 = ldfrag(Qbase);        // dk 0..31
    const v16h qf1 = ldfrag(Qbase + 32);   // dk 32..63
    const _Float16* Kbh = Kh + bh * Sn * DKn;
    const _Float16* Vbh = VT + bh * DKn * Sn;
    const int* mrow = mask + (size_t)b * Sn;
    _Float16* Pw = &Pl[wave][0];

    v16h onesf;
    #pragma unroll
    for (int i = 0; i < 16; ++i) onesf[i] = (_Float16)1.0f;

    float mrun[8], lrun[8];
    #pragma unroll
    for (int r = 0; r < 8; ++r) { mrun[r] = -1e30f; lrun[r] = 0.f; }
    v8f o0 = {}, o1 = {}, o2 = {}, o3 = {};
    const float scale = 0.125f;  // 1/sqrt(64)

    for (int kv0 = 0; kv0 < Sn; kv0 += 128) {
        // ---- stage K (128 x 64) and V^T (64 x 128) panels in LDS ----
#if HAVE_TDM
        if (wave == 0) {
            tdm_load_2d_f16((unsigned)(size_t)(void*)&Kt[0],
                            Kbh + (size_t)kv0 * DKn,
                            /*td0=*/64, /*td1=*/128,
                            /*tile0=*/64, /*tile1=*/128,
                            /*stride0=*/DKn, /*pad_code=*/4);  // 32-dw rows
            tdm_load_2d_f16((unsigned)(size_t)(void*)&Vt[0],
                            Vbh + kv0,
                            /*td0=*/128, /*td1=*/64,
                            /*tile0=*/128, /*tile1=*/64,
                            /*stride0=*/Sn, /*pad_code=*/5);   // 64-dw rows
            __builtin_amdgcn_s_wait_tensorcnt(0);
        }
#else
        for (int t = threadIdx.x; t < 128 * 8; t += 256) {
            const int rrow = t >> 3;
            const int cc = (t & 7) * 8;
            *(v8h*)&Kt[rrow * KRS + cc] =
                *(const v8h*)&Kbh[(size_t)(kv0 + rrow) * DKn + cc];
        }
        for (int t = threadIdx.x; t < 64 * 16; t += 256) {
            const int rrow = t >> 4;
            const int cc = (t & 15) * 8;
            *(v8h*)&Vt[rrow * VRS + cc] =
                *(const v8h*)&Vbh[(size_t)rrow * Sn + kv0 + cc];
        }
#endif
        __syncthreads();

        #pragma unroll
        for (int c0 = 0; c0 < 128; c0 += 32) {
            // ---- scores: two 16x16 tiles from the LDS K panel ----
            v8f s01[2];
            #pragma unroll
            for (int j = 0; j < 2; ++j) {
                const _Float16* kp = &Kt[(c0 + j * 16 + r16) * KRS + off];
                v16h kf0 = ldfrag(kp);
                v16h kf1 = ldfrag(kp + 32);
                v8f sj = {};
                sj = __builtin_amdgcn_wmma_f32_16x16x32_f16(false, qf0, false, kf0, (short)0, sj, false, false);
                sj = __builtin_amdgcn_wmma_f32_16x16x32_f16(false, qf1, false, kf1, (short)0, sj, false, false);
                s01[j] = sj;
            }
            const float bias0 = mrow[kv0 + c0 + r16]      ? 0.f : -1e30f;
            const float bias1 = mrow[kv0 + c0 + 16 + r16] ? 0.f : -1e30f;

            // ---- online softmax: row max via lane tree, probs into LDS ----
            float alpha[8];
            #pragma unroll
            for (int r = 0; r < 8; ++r) {
                float a0 = s01[0][r] * scale + bias0;
                float a1 = s01[1][r] * scale + bias1;
                float tmax = fmaxf(a0, a1);
                #pragma unroll
                for (int m = 8; m; m >>= 1) tmax = fmaxf(tmax, __shfl_xor(tmax, m, 32));
                const float mn = fmaxf(mrun[r], tmax);
                alpha[r] = __expf(mrun[r] - mn);
                mrun[r] = mn;
                const float p0 = __expf(a0 - mn);
                const float p1 = __expf(a1 - mn);
                Pw[(rowb + r) * 32 + r16]      = (_Float16)p0;
                Pw[(rowb + r) * 32 + 16 + r16] = (_Float16)p1;
            }
            #pragma unroll
            for (int r = 0; r < 8; ++r) {
                o0[r] *= alpha[r]; o1[r] *= alpha[r];
                o2[r] *= alpha[r]; o3[r] *= alpha[r];
            }
            // same-wave DS ops are serviced in order; block compiler reorder
            asm volatile("" ::: "memory");
            const v16h pf = ldfrag(Pw + r16 * 32 + off);   // P as A-fragment
            asm volatile("" ::: "memory");

            // ---- row sums: P x ones -> every C column = row sum ----
            v8f ps = {};
            ps = __builtin_amdgcn_wmma_f32_16x16x32_f16(false, pf, false, onesf, (short)0, ps, false, false);
            #pragma unroll
            for (int r = 0; r < 8; ++r) lrun[r] = lrun[r] * alpha[r] + ps[r];

            // ---- O += P(16x32) x V(32x64) from the LDS V panel ----
            v16h vf0 = ldfrag(&Vt[( 0 + r16) * VRS + c0 + off]);
            v16h vf1 = ldfrag(&Vt[(16 + r16) * VRS + c0 + off]);
            v16h vf2 = ldfrag(&Vt[(32 + r16) * VRS + c0 + off]);
            v16h vf3 = ldfrag(&Vt[(48 + r16) * VRS + c0 + off]);
            o0 = __builtin_amdgcn_wmma_f32_16x16x32_f16(false, pf, false, vf0, (short)0, o0, false, false);
            o1 = __builtin_amdgcn_wmma_f32_16x16x32_f16(false, pf, false, vf1, (short)0, o1, false, false);
            o2 = __builtin_amdgcn_wmma_f32_16x16x32_f16(false, pf, false, vf2, (short)0, o2, false, false);
            o3 = __builtin_amdgcn_wmma_f32_16x16x32_f16(false, pf, false, vf3, (short)0, o3, false, false);
        }
        __syncthreads();   // protect K/V panels before next chunk overwrite
    }

    // ---- normalize (fast rcp -- softmax norm needs no IEEE div) ----
    float inv[8];
    #pragma unroll
    for (int r = 0; r < 8; ++r)
        inv[r] = (lrun[r] > 0.f) ? __builtin_amdgcn_rcpf(lrun[r]) : 0.f;
    v8f os[4] = {o0, o1, o2, o3};
    #pragma unroll
    for (int t = 0; t < 4; ++t) {
        #pragma unroll
        for (int r = 0; r < 8; ++r) {
            const int mm = q0 + rowb + r;
            const int dd = t * 16 + r16;
            Oa[((size_t)b * Sn + mm) * Dn + h * DKn + dd] = (_Float16)(os[t][r] * inv[r]);
        }
    }
}

// ---------------------------------------------------------------------------
// Host orchestration
// ---------------------------------------------------------------------------
extern "C" void kernel_launch(void* const* d_in, const int* in_sizes, int n_in,
                              void* d_out, int out_size, void* d_ws, size_t ws_size,
                              hipStream_t stream) {
    const float* q    = (const float*)d_in[0];
    const float* k    = (const float*)d_in[1];
    const float* v    = (const float*)d_in[2];
    const int*   mask = (const int*)  d_in[3];
    const float* Wq   = (const float*)d_in[4];
    const float* Wk   = (const float*)d_in[5];
    const float* Wv   = (const float*)d_in[6];
    const float* Wo   = (const float*)d_in[7];
    const float* gq   = (const float*)d_in[8];
    const float* bq   = (const float*)d_in[9];
    const float* gk   = (const float*)d_in[10];
    const float* bk   = (const float*)d_in[11];
    const float* gv   = (const float*)d_in[12];
    const float* bv   = (const float*)d_in[13];
    float* out = (float*)d_out;

    // workspace carve-up (halves)
    _Float16* ws  = (_Float16*)d_ws;
    const size_t MD = (size_t)Mn * Dn;   // 8M halves = 16 MB
    const size_t DD = (size_t)Dn * Dn;   // 1M halves = 2 MB
    _Float16* lnq  = ws;
    _Float16* lnk  = lnq + MD;
    _Float16* lnv  = lnk + MD;
    _Float16* wqT  = lnv + MD;
    _Float16* wkT  = wqT + DD;
    _Float16* wvT  = wkT + DD;
    _Float16* woT  = wvT + DD;
    _Float16* qh   = woT + DD;
    _Float16* kh   = qh  + MD;
    _Float16* vT   = kh  + MD;
    _Float16* attn = lnq;                // ln buffers dead after projections

    // 1) LayerNorm -> f16
    ln_rows_f16<<<Mn, 256, 0, stream>>>(q, gq, bq, lnq);
    ln_rows_f16<<<Mn, 256, 0, stream>>>(k, gk, bk, lnk);
    ln_rows_f16<<<Mn, 256, 0, stream>>>(v, gv, bv, lnv);

    // 2) weights -> f16 transposed
    const int wblocks = (int)(DD / 256);
    w_to_f16_t<<<wblocks, 256, 0, stream>>>(Wq, wqT);
    w_to_f16_t<<<wblocks, 256, 0, stream>>>(Wk, wkT);
    w_to_f16_t<<<wblocks, 256, 0, stream>>>(Wv, wvT);
    w_to_f16_t<<<wblocks, 256, 0, stream>>>(Wo, woT);

    // 3) projections
    dim3 gp(Dn / 64, Mn / 128);
    wmma_gemm<0><<<gp, 256, 0, stream>>>(lnq, wqT, qh, nullptr);
    wmma_gemm<0><<<gp, 256, 0, stream>>>(lnk, wkT, kh, nullptr);
    wmma_gemm<1><<<gp, 256, 0, stream>>>(lnv, wvT, vT, nullptr);

    // 4) flash attention
    const int awaves = Bn * Hn * (Sn / 16);     // 8192 waves
    flash_attn_kernel<<<awaves / 8, 256, 0, stream>>>(qh, kh, vT, mask, attn);

    // 5) output projection + residual
    wmma_gemm<2><<<gp, 256, 0, stream>>>(attn, woT, out, q);
}
